// LTX2AudioAttnBlock_52733608461055
// MI455X (gfx1250) — matrix operand order repaired and checked by reading, not compiled, not run on hardware
//
#include <hip/hip_runtime.h>
#include <hip/hip_bf16.h>

// ---------------------------------------------------------------------------
// LTX2AudioAttnBlock for MI455X (gfx1250): GN -> QKV -> QK^T -> softmax ->
// PV -> proj + residual.  All GEMMs via v_wmma_f32_16x16x32_bf16 (wave32),
// 2x2 register blocking per wave (16 FLOP/B from cache) plus a 2x4 wave->tile
// swizzle inside each block so co-resident waves share operand rows through
// the WGP cache (cuts L2 traffic ~2.5x on the two N^2 GEMMs).
// Softmax runs row-wise over a 16x4096 f32 panel held in 257KB of LDS.
// B=2, C=512, H=W=64, N=4096, GROUPS=32.
// ---------------------------------------------------------------------------

typedef __attribute__((ext_vector_type(16))) __bf16 v16bf;
typedef __attribute__((ext_vector_type(8)))  __bf16 v8bf;
typedef __attribute__((ext_vector_type(8)))  float  v8f;

static constexpr int Bb = 2;
static constexpr int Cc = 512;
static constexpr int Nn = 4096;     // H*W
static constexpr int Gg = 32;
static constexpr float kQScale = 0.044194173824159223f; // 512^-0.5

// ---- WMMA fragment loaders (contiguous k = 16*(lane/16) + s for both A and B) ----------
__device__ __forceinline__ v16bf load_a_frag(const __bf16* base, int stride, int lane) {
  const __bf16* p = base + (size_t)(lane & 15) * stride + ((lane >> 4) << 4);
  return *(const v16bf*)p;
}

__device__ __forceinline__ v16bf load_b_frag(const __bf16* base, int stride, int lane) {
  const __bf16* p = base + (size_t)(lane & 15) * stride + ((lane >> 4) << 4);
  return *(const v16bf*)p;
}

__device__ __forceinline__ v8f wmma_bf16(v16bf a, v16bf b, v8f c) {
  return __builtin_amdgcn_wmma_f32_16x16x32_bf16(false, a, false, b, (short)0, c,
                                                 false, false);
}

__device__ __forceinline__ v8f zero8() {
  v8f z = {0.f, 0.f, 0.f, 0.f, 0.f, 0.f, 0.f, 0.f};
  return z;
}

// ---------------------------------------------------------------------------
// Kernel 0: convert the four CxC weight matrices to bf16 (row-major [o][c]).
// ---------------------------------------------------------------------------
__global__ __launch_bounds__(256) void cvt_w_kernel(const float* __restrict__ wq,
                                                    const float* __restrict__ wk,
                                                    const float* __restrict__ wv,
                                                    const float* __restrict__ wp,
                                                    __bf16* __restrict__ dst) {
  int i = blockIdx.x * 256 + threadIdx.x;            // 0 .. 4*C*C-1
  int m = i >> 18;                                   // C*C = 262144
  int off = i & (Cc * Cc - 1);
  const float* src = (m == 0) ? wq : (m == 1) ? wk : (m == 2) ? wv : wp;
  dst[i] = (__bf16)src[off];
}

// ---------------------------------------------------------------------------
// Kernel 1: GroupNorm, writes hT pixel-major [B][N][C] bf16.
// ---------------------------------------------------------------------------
__global__ __launch_bounds__(256) void gn_kernel(const float* __restrict__ x,
                                                 const float* __restrict__ gamma,
                                                 const float* __restrict__ beta,
                                                 __bf16* __restrict__ hT) {
  const int b = blockIdx.x / Gg;
  const int g = blockIdx.x % Gg;
  const int c0 = g * 16;
  const float* xb = x + ((size_t)b * Cc + c0) * Nn;  // 16*4096 contiguous floats
  const int w = threadIdx.x >> 5, lane = threadIdx.x & 31;

  float s = 0.f, s2 = 0.f;
#pragma unroll 4
  for (int it = 0; it < 256; ++it) {
    float v = xb[it * 256 + threadIdx.x];
    s += v; s2 += v * v;
  }
#pragma unroll
  for (int m = 16; m >= 1; m >>= 1) {
    s  += __shfl_xor(s,  m, 32);
    s2 += __shfl_xor(s2, m, 32);
  }
  __shared__ float rs[8], rs2[8], stats[2];
  if (lane == 0) { rs[w] = s; rs2[w] = s2; }
  __syncthreads();
  if (threadIdx.x == 0) {
    float S = 0.f, S2 = 0.f;
#pragma unroll
    for (int i = 0; i < 8; ++i) { S += rs[i]; S2 += rs2[i]; }
    float mean = S * (1.f / 65536.f);
    float var  = S2 * (1.f / 65536.f) - mean * mean;
    stats[0] = mean;
    stats[1] = rsqrtf(var + 1e-6f);
  }
  __syncthreads();
  const float mean = stats[0], inv = stats[1];

#pragma unroll 4
  for (int it = 0; it < 256; ++it) {
    int idx = it * 256 + threadIdx.x;
    int c = idx >> 12;            // 0..15
    int n = idx & (Nn - 1);
    float v = (xb[idx] - mean) * inv * gamma[c0 + c] + beta[c0 + c];
    hT[((size_t)b * Nn + n) * Cc + c0 + c] = (__bf16)v;
  }
}

// ---------------------------------------------------------------------------
// Kernel 2: QKV projections.  32x32 super-tile per wave (2x2 WMMA blocking);
// each block covers 64 out-chan x 128 pixels (2x4 wave swizzle) so waves on
// the same WGP share hT rows via WGP$.
//  mat 0/1 (q,k): M=out-chan, N=pixel -> contiguous stores into qT/kT [B,N,C]
//  mat 2   (v)  : M=pixel, N=out-chan -> contiguous stores into v   [B,C,N]
// ---------------------------------------------------------------------------
__global__ __launch_bounds__(256) void qkv_kernel(const __bf16* __restrict__ hT,
                                                  const __bf16* __restrict__ wbf,
                                                  const float* __restrict__ bq,
                                                  const float* __restrict__ bk,
                                                  const float* __restrict__ bv,
                                                  __bf16* __restrict__ qT,
                                                  __bf16* __restrict__ kT,
                                                  __bf16* __restrict__ vv) {
  const int wv8  = threadIdx.x >> 5;                // wave in block, 0..7
  const int lane = threadIdx.x & 31;
  int mat = blockIdx.x % 3; int t = blockIdx.x / 3; // 1536 blocks
  int ob  = t % 8;  t /= 8;                         // 64-och block
  int pb  = t % 32; int b = t / 32;                 // 128-pixel block
  const int oct2 = ob * 2 + (wv8 >> 2);             // 32-och super-tile
  const int pxt2 = pb * 4 + (wv8 & 3);              // 32-pixel super-tile

  const __bf16* w  = wbf + (size_t)mat * Cc * Cc;
  const float*  bs = (mat == 0) ? bq : (mat == 1) ? bk : bv;
  const __bf16* wr = w  + (size_t)(oct2 * 32) * Cc;
  const __bf16* hr = hT + ((size_t)b * Nn + pxt2 * 32) * Cc;

  v8f c00 = zero8(), c01 = zero8(), c10 = zero8(), c11 = zero8();

  if (mat < 2) {  // A = weight rows, B = hT rows
#pragma unroll
    for (int kk = 0; kk < 16; ++kk) {
      v16bf a0 = load_a_frag(wr + kk * 32, Cc, lane);
      v16bf a1 = load_a_frag(wr + (size_t)16 * Cc + kk * 32, Cc, lane);
      v16bf b0 = load_b_frag(hr + kk * 32, Cc, lane);
      v16bf b1 = load_b_frag(hr + (size_t)16 * Cc + kk * 32, Cc, lane);
      c00 = wmma_bf16(a0, b0, c00);
      c01 = wmma_bf16(a0, b1, c01);
      c10 = wmma_bf16(a1, b0, c10);
      c11 = wmma_bf16(a1, b1, c11);
    }
    __bf16* dstm = (mat == 0) ? qT : kT;
#pragma unroll
    for (int ti = 0; ti < 2; ++ti) {
#pragma unroll
      for (int tj = 0; tj < 2; ++tj) {
        v8f acc = (ti == 0) ? ((tj == 0) ? c00 : c01) : ((tj == 0) ? c10 : c11);
        int pix  = pxt2 * 32 + tj * 16 + (lane & 15);
        int och0 = oct2 * 32 + ti * 16 + ((lane >> 4) << 3);
        v8bf o;
#pragma unroll
        for (int r = 0; r < 8; ++r) {
          float f = acc[r] + bs[och0 + r];
          if (mat == 0) f *= kQScale;           // fold C^-0.5 into q
          o[r] = (__bf16)f;
        }
        *(v8bf*)(dstm + ((size_t)b * Nn + pix) * Cc + och0) = o;
      }
    }
  } else {        // A = hT rows (pix), B = weight rows (och)
#pragma unroll
    for (int kk = 0; kk < 16; ++kk) {
      v16bf a0 = load_a_frag(hr + kk * 32, Cc, lane);
      v16bf a1 = load_a_frag(hr + (size_t)16 * Cc + kk * 32, Cc, lane);
      v16bf b0 = load_b_frag(wr + kk * 32, Cc, lane);
      v16bf b1 = load_b_frag(wr + (size_t)16 * Cc + kk * 32, Cc, lane);
      c00 = wmma_bf16(a0, b0, c00);
      c01 = wmma_bf16(a0, b1, c01);
      c10 = wmma_bf16(a1, b0, c10);
      c11 = wmma_bf16(a1, b1, c11);
    }
#pragma unroll
    for (int ti = 0; ti < 2; ++ti) {          // pix tile
#pragma unroll
      for (int tj = 0; tj < 2; ++tj) {        // och tile
        v8f acc = (ti == 0) ? ((tj == 0) ? c00 : c01) : ((tj == 0) ? c10 : c11);
        int och  = oct2 * 32 + tj * 16 + (lane & 15);
        int pix0 = pxt2 * 32 + ti * 16 + ((lane >> 4) << 3);
        float bias = bs[och];
        v8bf o;
#pragma unroll
        for (int r = 0; r < 8; ++r) o[r] = (__bf16)(acc[r] + bias);
        *(v8bf*)(vv + ((size_t)b * Cc + och) * Nn + pix0) = o;
      }
    }
  }
}

// ---------------------------------------------------------------------------
// Kernel 3: scores GEMM, logits[q][k] = qT[q] . kT[k]  (q pre-scaled).
// Orientation M=key, N=query so each lane stores 8 contiguous keys (bf16).
// Block covers 64 keys x 128 queries (2x4 wave swizzle -> WGP$ sharing).
// ---------------------------------------------------------------------------
__global__ __launch_bounds__(256) void scores_gemm_kernel(const __bf16* __restrict__ qT,
                                                          const __bf16* __restrict__ kT,
                                                          __bf16* __restrict__ attn) {
  const int wv8  = threadIdx.x >> 5;
  const int lane = threadIdx.x & 31;
  int kb = blockIdx.x & 63; int t = blockIdx.x >> 6; // 4096 blocks
  int qb = t & 31; int b = t >> 5;
  const int kt2 = kb * 2 + (wv8 >> 2);               // 32-key super-tile
  const int qt2 = qb * 4 + (wv8 & 3);                // 32-query super-tile

  const __bf16* kr = kT + ((size_t)b * Nn + kt2 * 32) * Cc;  // A: key rows
  const __bf16* qr = qT + ((size_t)b * Nn + qt2 * 32) * Cc;  // B: query rows

  v8f c00 = zero8(), c01 = zero8(), c10 = zero8(), c11 = zero8();
#pragma unroll
  for (int kk = 0; kk < 16; ++kk) {
    v16bf a0 = load_a_frag(kr + kk * 32, Cc, lane);
    v16bf a1 = load_a_frag(kr + (size_t)16 * Cc + kk * 32, Cc, lane);
    v16bf b0 = load_b_frag(qr + kk * 32, Cc, lane);
    v16bf b1 = load_b_frag(qr + (size_t)16 * Cc + kk * 32, Cc, lane);
    c00 = wmma_bf16(a0, b0, c00);
    c01 = wmma_bf16(a0, b1, c01);
    c10 = wmma_bf16(a1, b0, c10);
    c11 = wmma_bf16(a1, b1, c11);
  }
#pragma unroll
  for (int ti = 0; ti < 2; ++ti) {          // key tile
#pragma unroll
    for (int tj = 0; tj < 2; ++tj) {        // query tile
      v8f acc = (ti == 0) ? ((tj == 0) ? c00 : c01) : ((tj == 0) ? c10 : c11);
      int q  = qt2 * 32 + tj * 16 + (lane & 15);
      int k0 = kt2 * 32 + ti * 16 + ((lane >> 4) << 3);
      v8bf o;
#pragma unroll
      for (int r = 0; r < 8; ++r) o[r] = (__bf16)acc[r];
      *(v8bf*)(attn + ((size_t)b * Nn + q) * Nn + k0) = o;
    }
  }
}

// ---------------------------------------------------------------------------
// Kernel 4: in-place row softmax over bf16 logits.  One WG per 16 rows; the
// 16x4096 f32 panel lives in 257KB of LDS (CDNA5: 320KB/WGP).  Rows are
// disjoint across WGs, so normalizing in place is safe.
// ---------------------------------------------------------------------------
__global__ __launch_bounds__(256) void softmax_kernel(__bf16* __restrict__ attn) {
  __shared__ float sc[16][Nn + 16];                 // padded rows
  const int b  = blockIdx.x >> 8;
  const int rt = blockIdx.x & 255;
  const int w = threadIdx.x >> 5, lane = threadIdx.x & 31;
  __bf16* base = attn + ((size_t)b * Nn + rt * 16) * Nn;

  // load 16 rows of bf16 logits -> f32 LDS (8 elems per thread per iter)
#pragma unroll 4
  for (int it = 0; it < 32; ++it) {
    int idx8 = (it * 256 + threadIdx.x) * 8;
    int row = idx8 >> 12;
    int col = idx8 & (Nn - 1);
    v8bf d = *(const v8bf*)(base + (size_t)row * Nn + col);
#pragma unroll
    for (int e = 0; e < 8; ++e) sc[row][col + e] = (float)d[e];
  }
  __syncthreads();

  // rows 2w and 2w+1: max, exp-sum, normalize, write bf16 back in place
#pragma unroll
  for (int rr = 0; rr < 2; ++rr) {
    int row = w * 2 + rr;
    float mx = -3.4e38f;
    for (int j = lane; j < Nn; j += 32) mx = fmaxf(mx, sc[row][j]);
#pragma unroll
    for (int m = 16; m >= 1; m >>= 1) mx = fmaxf(mx, __shfl_xor(mx, m, 32));
    float sum = 0.f;
    for (int j = lane; j < Nn; j += 32) {
      float e = __expf(sc[row][j] - mx);
      sc[row][j] = e;
      sum += e;
    }
#pragma unroll
    for (int m = 16; m >= 1; m >>= 1) sum += __shfl_xor(sum, m, 32);
    float rinv = 1.f / sum;
    __bf16* arow = base + (size_t)row * Nn;
    for (int j = lane; j < Nn; j += 32) arow[j] = (__bf16)(sc[row][j] * rinv);
  }
}

// ---------------------------------------------------------------------------
// Kernel 5: O = attn * V^T, oriented (M=channel, N=query), 2x2 blocking.
// Block covers 128 channels x 64 queries (4x2 wave swizzle -> WGP$ sharing
// of both attn and v rows).  Output oT [B][N][C] bf16.
// ---------------------------------------------------------------------------
__global__ __launch_bounds__(256) void pv_kernel(const __bf16* __restrict__ attn,
                                                 const __bf16* __restrict__ vv,
                                                 __bf16* __restrict__ oT) {
  const int wv8  = threadIdx.x >> 5;
  const int lane = threadIdx.x & 31;
  int cb = blockIdx.x & 3; int t = blockIdx.x >> 2;  // 512 blocks
  int qb = t & 63; int b = t >> 6;
  const int ct2 = cb * 4 + (wv8 & 3);                // 32-channel super-tile
  const int qt2 = qb * 2 + (wv8 >> 2);               // 32-query super-tile

  const __bf16* ab = attn + ((size_t)b * Nn + qt2 * 32) * Nn;
  const __bf16* vb = vv   + ((size_t)b * Cc + ct2 * 32) * Nn;

  v8f c00 = zero8(), c01 = zero8(), c10 = zero8(), c11 = zero8();
#pragma unroll 2
  for (int jt = 0; jt < 128; ++jt) {
    v16bf a0 = load_a_frag(vb + jt * 32, Nn, lane);                    // ch tile 0
    v16bf a1 = load_a_frag(vb + (size_t)16 * Nn + jt * 32, Nn, lane);  // ch tile 1
    v16bf b0 = load_b_frag(ab + jt * 32, Nn, lane);                    // q tile 0
    v16bf b1 = load_b_frag(ab + (size_t)16 * Nn + jt * 32, Nn, lane);  // q tile 1
    c00 = wmma_bf16(a0, b0, c00);
    c01 = wmma_bf16(a0, b1, c01);
    c10 = wmma_bf16(a1, b0, c10);
    c11 = wmma_bf16(a1, b1, c11);
  }
#pragma unroll
  for (int ti = 0; ti < 2; ++ti) {          // channel tile
#pragma unroll
    for (int tj = 0; tj < 2; ++tj) {        // query tile
      v8f acc = (ti == 0) ? ((tj == 0) ? c00 : c01) : ((tj == 0) ? c10 : c11);
      int q   = qt2 * 32 + tj * 16 + (lane & 15);
      int ch0 = ct2 * 32 + ti * 16 + ((lane >> 4) << 3);
      v8bf o;
#pragma unroll
      for (int r = 0; r < 8; ++r) o[r] = (__bf16)acc[r];
      *(v8bf*)(oT + ((size_t)b * Nn + q) * Cc + ch0) = o;
    }
  }
}

// ---------------------------------------------------------------------------
// Kernel 6: out = x + wp*o + bp, oriented (M=pixel, N=out-chan), 2x2 blocking,
// 2x4 wave swizzle.  Each lane stores 8 consecutive f32 of the final output.
// ---------------------------------------------------------------------------
__global__ __launch_bounds__(256) void proj_kernel(const __bf16* __restrict__ oT,
                                                   const __bf16* __restrict__ wpb,
                                                   const float* __restrict__ bp,
                                                   const float* __restrict__ x,
                                                   float* __restrict__ out) {
  const int wv8  = threadIdx.x >> 5;
  const int lane = threadIdx.x & 31;
  int ob2 = blockIdx.x & 7; int t = blockIdx.x >> 3; // 512 blocks
  int pb  = t & 31; int b = t >> 5;
  const int oct2 = ob2 * 2 + (wv8 >> 2);             // 32-out-chan super-tile
  const int pxt2 = pb * 4 + (wv8 & 3);               // 32-pixel super-tile

  const __bf16* obp = oT  + ((size_t)b * Nn + pxt2 * 32) * Cc;
  const __bf16* wr  = wpb + (size_t)(oct2 * 32) * Cc;

  v8f c00 = zero8(), c01 = zero8(), c10 = zero8(), c11 = zero8();
#pragma unroll
  for (int kk = 0; kk < 16; ++kk) {
    v16bf a0 = load_a_frag(obp + kk * 32, Cc, lane);                    // pix tile 0
    v16bf a1 = load_a_frag(obp + (size_t)16 * Cc + kk * 32, Cc, lane);  // pix tile 1
    v16bf b0 = load_b_frag(wr + kk * 32, Cc, lane);                     // och tile 0
    v16bf b1 = load_b_frag(wr + (size_t)16 * Cc + kk * 32, Cc, lane);   // och tile 1
    c00 = wmma_bf16(a0, b0, c00);
    c01 = wmma_bf16(a0, b1, c01);
    c10 = wmma_bf16(a1, b0, c10);
    c11 = wmma_bf16(a1, b1, c11);
  }
#pragma unroll
  for (int ti = 0; ti < 2; ++ti) {          // pixel tile
#pragma unroll
    for (int tj = 0; tj < 2; ++tj) {        // out-channel tile
      v8f acc = (ti == 0) ? ((tj == 0) ? c00 : c01) : ((tj == 0) ? c10 : c11);
      int och  = oct2 * 32 + tj * 16 + (lane & 15);
      int pix0 = pxt2 * 32 + ti * 16 + ((lane >> 4) << 3);
      float bias = bp[och];
      size_t base = ((size_t)b * Cc + och) * Nn + pix0;
#pragma unroll
      for (int r = 0; r < 8; ++r) out[base + r] = x[base + r] + acc[r] + bias;
    }
  }
}

// ---------------------------------------------------------------------------
extern "C" void kernel_launch(void* const* d_in, const int* in_sizes, int n_in,
                              void* d_out, int out_size, void* d_ws, size_t ws_size,
                              hipStream_t stream) {
  const float* x     = (const float*)d_in[0];
  const float* gamma = (const float*)d_in[1];
  const float* beta  = (const float*)d_in[2];
  const float* wq    = (const float*)d_in[3];
  const float* bq    = (const float*)d_in[4];
  const float* wk    = (const float*)d_in[5];
  const float* bk    = (const float*)d_in[6];
  const float* wv    = (const float*)d_in[7];
  const float* bv    = (const float*)d_in[8];
  const float* wp    = (const float*)d_in[9];
  const float* bp    = (const float*)d_in[10];
  float* out = (float*)d_out;

  // workspace carve-up (all 256B-aligned sizes; ~110 MB total)
  char* p = (char*)d_ws;
  __bf16* wbf  = (__bf16*)p; p += (size_t)4 * Cc * Cc * 2;        //   2 MB
  __bf16* hT   = (__bf16*)p; p += (size_t)Bb * Nn * Cc * 2;       //   8 MB
  __bf16* qT   = (__bf16*)p; p += (size_t)Bb * Nn * Cc * 2;       //   8 MB
  __bf16* kT   = (__bf16*)p; p += (size_t)Bb * Nn * Cc * 2;       //   8 MB
  __bf16* vv   = (__bf16*)p; p += (size_t)Bb * Cc * Nn * 2;       //   8 MB
  __bf16* oT   = (__bf16*)p; p += (size_t)Bb * Nn * Cc * 2;       //   8 MB
  __bf16* attn = (__bf16*)p; p += (size_t)Bb * Nn * Nn * 2;       //  64 MB (logits -> probs in place)

  cvt_w_kernel      <<<4096, 256, 0, stream>>>(wq, wk, wv, wp, wbf);
  gn_kernel         <<<Bb * Gg, 256, 0, stream>>>(x, gamma, beta, hT);
  qkv_kernel        <<<1536, 256, 0, stream>>>(hT, wbf, bq, bk, bv, qT, kT, vv);
  scores_gemm_kernel<<<4096, 256, 0, stream>>>(qT, kT, attn);
  softmax_kernel    <<<Bb * (Nn / 16), 256, 0, stream>>>(attn);
  pv_kernel         <<<512, 256, 0, stream>>>(attn, vv, oT);
  proj_kernel       <<<512, 256, 0, stream>>>(oT, wbf + (size_t)3 * Cc * Cc, bp, x, out);
}
